// GATLayer_84825604096092
// MI455X (gfx1250) — matrix operand order repaired and checked by reading, not compiled, not run on hardware
//
#include <hip/hip_runtime.h>
#include <hip/hip_bf16.h>

// GAT layer fused kernel set for MI455X (gfx1250, wave32, WMMA).
//
// Inputs (setup_inputs order): h [8192,256] f32, adj [8192,8192] i32,
//                              W [256,64] f32, a [128,1] f32
// Output: elu(softmax(mask(leakyrelu(f_dst_i + f_src_j))) @ Wh)  [8192,64] f32
//
// Single pass over adj (268 MB -> ~11.5us @ 23.3 TB/s HBM) using the
// shift-invariant softmax bound c_i = leakyrelu(f_dst[i] + max_j f_src[j]).

typedef __attribute__((ext_vector_type(16))) __bf16 v16bf16;
typedef __attribute__((ext_vector_type(8)))  float  v8f;

#define N_NODES 8192
#define F_IN    256
#define F_OUT   64
#define ALPHA   0.2f
#define LOG2E   1.44269504088896340736f

__device__ __forceinline__ float lrelu(float x) { return x > 0.0f ? x : ALPHA * x; }

// ---------------------------------------------------------------------------
// Kernel A: Wh = h @ W  (stored transposed as bf16, WhT[64][8192]),
//           f_src[i] = Wh[i,:] @ a[:64],  f_dst[i] = Wh[i,:] @ a[64:]
// Block: 256 threads = 4 rows x 64 output features. Grid: 2048.
// ---------------------------------------------------------------------------
__global__ __launch_bounds__(256)
void gat_proj_kernel(const float* __restrict__ h,
                     const float* __restrict__ W,
                     const float* __restrict__ a,
                     __bf16* __restrict__ whT,
                     float*  __restrict__ f_src,
                     float*  __restrict__ f_dst)
{
    __shared__ float s_src[4][64];
    __shared__ float s_dst[4][64];

    const int tid = threadIdx.x;
    const int n   = tid & 63;        // output feature
    const int rl  = tid >> 6;        // local row 0..3
    const int i   = blockIdx.x * 4 + rl;

    const float* hrow = h + (size_t)i * F_IN;
    float acc = 0.0f;
#pragma unroll 8
    for (int k = 0; k < F_IN; ++k)
        acc = fmaf(hrow[k], W[k * F_OUT + n], acc);

    // bf16 transposed copy for the WMMA B operand (L2-resident, 1 MB)
    whT[(size_t)n * N_NODES + i] = (__bf16)acc;

    s_src[rl][n] = acc * a[n];
    s_dst[rl][n] = acc * a[F_OUT + n];
    __syncthreads();

    if (n == 0) {
        float ss = 0.0f, sd = 0.0f;
#pragma unroll
        for (int c = 0; c < 64; ++c) { ss += s_src[rl][c]; sd += s_dst[rl][c]; }
        f_src[i] = ss;
        f_dst[i] = sd;
    }
}

// ---------------------------------------------------------------------------
// Kernel C: global max of f_src (8192 values) -> M (softmax shift bound)
// ---------------------------------------------------------------------------
__global__ __launch_bounds__(256)
void fsrc_max_kernel(const float* __restrict__ f_src, float* __restrict__ Mout)
{
    __shared__ float sm[256];
    float m = -3.4e38f;
    for (int i = threadIdx.x; i < N_NODES; i += 256)
        m = fmaxf(m, f_src[i]);
    sm[threadIdx.x] = m;
    __syncthreads();
    for (int s = 128; s > 0; s >>= 1) {
        if (threadIdx.x < s) sm[threadIdx.x] = fmaxf(sm[threadIdx.x], sm[threadIdx.x + s]);
        __syncthreads();
    }
    if (threadIdx.x == 0) Mout[0] = sm[0];
}

// ---------------------------------------------------------------------------
// Kernel B (main): for each 16-row block, stream adj once, build P tiles in
// WMMA A layout, accumulate P @ Wh with v_wmma_f32_16x16x32_bf16, normalize
// by the row sum, apply elu.
// Grid: 512 blocks x 128 threads (4 waves; wave w covers j in [w*2048, ...)).
// ---------------------------------------------------------------------------
__global__ __launch_bounds__(128)
void gat_attn_kernel(const int* __restrict__ adj,
                     const __bf16* __restrict__ whT,
                     const float* __restrict__ f_src,
                     const float* __restrict__ f_dst,
                     const float* __restrict__ Mptr,
                     float* __restrict__ out)
{
    __shared__ float ldsC[4][16][64];   // per-wave partial C tiles
    __shared__ float ldsS[4][16];       // per-wave partial row sums

    const int tid  = threadIdx.x;
    const int wave = tid >> 5;
    const int lane = tid & 31;
    const int row  = lane & 15;         // M index (and B-matrix N index)
    const int hi   = lane >> 4;         // lane half
    const int i0   = blockIdx.x * 16;   // global row base

    const float M      = Mptr[0];
    const float fd     = f_dst[i0 + row];
    const float cshift = lrelu(fd + M); // >= every row entry -> exp(arg) <= 1

    v8f cacc[4] = {v8f{}, v8f{}, v8f{}, v8f{}};
    float rsum = 0.0f;

    const int    kbA    = hi * 16;      // A-layout K base for this lane half
    const int    kbB    = hi * 16;      // B-layout K base for this lane half
    const size_t adjRow = (size_t)(i0 + row) * N_NODES;

    const int jb0 = wave * 2048;
    const int jb1 = jb0 + 2048;

    for (int jb = jb0; jb < jb1; jb += 32) {
        // --- adjacency + source coefficients for this lane's 16 (row,K) slots
        const int4*   ap = (const int4*)(adj + adjRow + jb + kbA);
        const float4* sp = (const float4*)(f_src + jb + kbA);
        int4   a0 = ap[0], a1 = ap[1], a2 = ap[2], a3 = ap[3];   // K: kbA+{0..15}
        float4 s0 = sp[0], s1 = sp[1], s2 = sp[2], s3 = sp[3];

        // Speculative prefetch ~8 iterations ahead (safe if OOB: dropped).
        __builtin_prefetch(adj + adjRow + jb + 256, 0, 3);   // global_prefetch_b8

        // --- P tile element: p = adj ? exp(leakyrelu(fd+fs) - cshift) : 0
        // Branch-free: raw v_exp_f32 (arg <= 0, no fixup needed) times a
        // {0,1} mask -> v_cmp + v_cndmask + v_mul, no exec-mask churn.
        float pv[16];
        v16bf16 pa;
#define PEL(u, av, sv)                                                       \
        { float e_ = lrelu(fd + (sv));                                       \
          float m_ = ((av) > 0) ? 1.0f : 0.0f;                               \
          float p_ = m_ * __builtin_amdgcn_exp2f((e_ - cshift) * LOG2E);     \
          pv[u] = p_; pa[u] = (__bf16)p_; }
        PEL(0,  a0.x, s0.x)  PEL(1,  a0.y, s0.y)  PEL(2,  a0.z, s0.z)  PEL(3,  a0.w, s0.w)
        PEL(4,  a1.x, s1.x)  PEL(5,  a1.y, s1.y)  PEL(6,  a1.z, s1.z)  PEL(7,  a1.w, s1.w)
        PEL(8,  a2.x, s2.x)  PEL(9,  a2.y, s2.y)  PEL(10, a2.z, s2.z)  PEL(11, a2.w, s2.w)
        PEL(12, a3.x, s3.x)  PEL(13, a3.y, s3.y)  PEL(14, a3.z, s3.z)  PEL(15, a3.w, s3.w)
#undef PEL

        // Tree-sum the 16 probabilities (short dep chains, overlaps WMMA).
        float t01 = (pv[0] + pv[1]) + (pv[2] + pv[3]);
        float t23 = (pv[4] + pv[5]) + (pv[6] + pv[7]);
        float t45 = (pv[8] + pv[9]) + (pv[10] + pv[11]);
        float t67 = (pv[12] + pv[13]) + (pv[14] + pv[15]);
        rsum += (t01 + t23) + (t45 + t67);

        // --- 4 output tiles of 16 columns each: C[t] += P(16x32) * WhT tile
#pragma unroll
        for (int t = 0; t < 4; ++t) {
            const uint4* bp =
                (const uint4*)(whT + (size_t)(t * 16 + row) * N_NODES + jb + kbB);
            union { uint4 q[2]; v16bf16 v; } ub;
            ub.q[0] = bp[0];
            ub.q[1] = bp[1];
            cacc[t] = __builtin_amdgcn_wmma_f32_16x16x32_bf16(
                false, pa, false, ub.v, (short)0, cacc[t], false, false);
        }
    }

    // --- reduce row sums: lanes l and l^16 hold the same row
    float rtot = rsum + __shfl_xor(rsum, 16, 32);
    if (hi == 0) ldsS[wave][row] = rtot;

    // --- spill C tiles (C layout: VGPR v, lane<16 -> M=v, else M=v+8; N=row)
#pragma unroll
    for (int t = 0; t < 4; ++t)
#pragma unroll
        for (int v = 0; v < 8; ++v)
            ldsC[wave][v + hi * 8][t * 16 + row] = cacc[t][v];
    __syncthreads();

    // --- combine 4 waves, normalize, elu, store (128 thr x 8 elems = 16x64)
#pragma unroll
    for (int u = 0; u < 8; ++u) {
        int e = tid * 8 + u;
        int r = e >> 6, col = e & 63;
        float vsum = ldsC[0][r][col] + ldsC[1][r][col]
                   + ldsC[2][r][col] + ldsC[3][r][col];
        float rs = ldsS[0][r] + ldsS[1][r] + ldsS[2][r] + ldsS[3][r];
        rs = rs > 0.0f ? rs : 1.0f;          // fully-masked-row guard
        float hp = vsum / rs;
        out[(size_t)(i0 + r) * F_OUT + col] = hp > 0.0f ? hp : (expf(hp) - 1.0f);
    }
}

// ---------------------------------------------------------------------------
extern "C" void kernel_launch(void* const* d_in, const int* in_sizes, int n_in,
                              void* d_out, int out_size, void* d_ws, size_t ws_size,
                              hipStream_t stream)
{
    const float* h   = (const float*)d_in[0];
    const int*   adj = (const int*)  d_in[1];
    const float* W   = (const float*)d_in[2];
    const float* a   = (const float*)d_in[3];
    float* out = (float*)d_out;

    char* ws = (char*)d_ws;
    __bf16* whT   = (__bf16*)ws;                               // 64*8192*2 = 1 MB
    float*  f_src = (float*)(ws + (size_t)1048576);            // 32 KB
    float*  f_dst = (float*)(ws + (size_t)1048576 + 32768);    // 32 KB
    float*  Mbuf  = (float*)(ws + (size_t)1048576 + 65536);    // 4 B

    gat_proj_kernel<<<N_NODES / 4, 256, 0, stream>>>(h, W, a, whT, f_src, f_dst);
    fsrc_max_kernel<<<1, 256, 0, stream>>>(f_src, Mbuf);
    gat_attn_kernel<<<N_NODES / 16, 128, 0, stream>>>(adj, whT, f_src, f_dst, Mbuf, out);
}
